// DiTBlock_55972013802288
// MI455X (gfx1250) — matrix operand order repaired and checked by reading, not compiled, not run on hardware
//
#include <hip/hip_runtime.h>
#include <hip/hip_bf16.h>
#include <math.h>

typedef __bf16 bf16;
typedef __attribute__((ext_vector_type(16))) __bf16 v16bf;
typedef __attribute__((ext_vector_type(8)))  float  v8f;

struct alignas(16) U4 { unsigned int x, y, z, w; };
union FragB16 { U4 q[2]; v16bf v; };

// ---------------------------------------------------------------------------
// CDNA5 data-mover paths (guarded so absence degrades, not fails)
// ---------------------------------------------------------------------------
#if __has_builtin(__builtin_amdgcn_tensor_load_to_lds)
#define USE_TDM 1
#endif
#if __has_builtin(__builtin_amdgcn_global_load_async_to_lds_b128)
#define USE_ASYNC 1
#endif

__device__ __forceinline__ unsigned lds_off_of(const void* p) {
    // flat LDS-aperture address: low 32 bits are the LDS byte offset
    return (unsigned)(unsigned long long)(size_t)p;
}

#if defined(USE_TDM)
typedef unsigned int u32x4 __attribute__((ext_vector_type(4)));
typedef int          i32x4 __attribute__((ext_vector_type(4)));
typedef int          i32x8 __attribute__((ext_vector_type(8)));

// 2D bf16 tile DMA: tile_h rows of tile_w elements, row stride in elements.
// D# layout per CDNA5 ISA (group0 + group1, 2D form).
__device__ __forceinline__ void tdm_load_tile_bf16(unsigned lds_byte_off,
                                                   const void* gtile,
                                                   unsigned tile_w, unsigned tile_h,
                                                   unsigned long long stride_elems) {
    unsigned long long ga = (unsigned long long)(size_t)gtile;
    u32x4 g0;
    g0.x = 1u;                                            // count=1, user mode
    g0.y = lds_byte_off;                                  // lds_addr (bytes)
    g0.z = (unsigned)ga;                                  // global_addr[31:0]
    g0.w = (unsigned)((ga >> 32) & 0x01ffffffull) | (2u << 30); // [56:32] | type=2
    i32x8 g1;
    g1[0] = (int)(1u << 16);                              // wg_mask=0, data_size=1 (2B)
    g1[1] = (int)((tile_w & 0xffffu) << 16);              // tensor_dim0[15:0]
    g1[2] = (int)(((tile_w >> 16) & 0xffffu) | ((tile_h & 0xffffu) << 16)); // d0 hi | d1 lo
    g1[3] = (int)(((tile_h >> 16) & 0xffffu) | ((tile_w & 0xffffu) << 16)); // d1 hi | tile_dim0
    g1[4] = (int)(tile_h & 0xffffu);                      // tile_dim1 | tile_dim2=0
    g1[5] = (int)(stride_elems & 0xffffffffull);          // dim0_stride[31:0]
    g1[6] = (int)((stride_elems >> 32) & 0xffffull);      // dim0_stride[47:32] | d1s=0
    g1[7] = 0;
    i32x4 z4 = {0, 0, 0, 0};
#if __clang_major__ >= 23
    i32x8 z8 = {0, 0, 0, 0, 0, 0, 0, 0};
    __builtin_amdgcn_tensor_load_to_lds(g0, g1, z4, z4, z8, 0);
#else
    __builtin_amdgcn_tensor_load_to_lds(g0, g1, z4, z4, 0);
#endif
}
__device__ __forceinline__ void tdm_wait0() {
    __builtin_amdgcn_s_wait_tensorcnt(0);
}
#endif  // USE_TDM

#if defined(USE_ASYNC)
// builtin signature (from diagnostic): (v4i AS1*, v4i AS3*, imm i32, imm i32)
typedef int vsi4 __attribute__((vector_size(4 * sizeof(int))));
__device__ __forceinline__ void async_b128(const void* g, void* l) {
    __builtin_amdgcn_global_load_async_to_lds_b128(
        (__attribute__((address_space(1))) vsi4*)(size_t)g,
        (__attribute__((address_space(3))) vsi4*)(unsigned)(size_t)l,
        0, 0);
}
__device__ __forceinline__ void async_wait0() {
#if __has_builtin(__builtin_amdgcn_s_wait_asynccnt)
    __builtin_amdgcn_s_wait_asynccnt(0);
#else
    asm volatile("s_wait_asynccnt 0x0" ::: "memory");
#endif
}
#endif  // USE_ASYNC

__device__ __forceinline__ float mish_f(float v) {
    float sp = (v > 20.0f) ? v : log1pf(expf(v));
    return v * tanhf(sp);
}

__device__ __forceinline__ v8f zero_v8f() {
    v8f z;
#pragma unroll
    for (int r = 0; r < 8; ++r) z[r] = 0.0f;
    return z;
}

// ---------------------------------------------------------------------------
// Weight transpose + fp32 -> bf16 cast:  W[K][N] -> WT[N][K]
// ---------------------------------------------------------------------------
__global__ void transpose_cast_k(const float* __restrict__ W, bf16* __restrict__ WT,
                                 int K, int N) {
    size_t i = (size_t)blockIdx.x * blockDim.x + threadIdx.x;
    size_t total = (size_t)K * N;
    if (i < total) {
        size_t n = i / K, k = i % K;
        WT[i] = (bf16)W[k * (size_t)N + n];
    }
}

// ---------------------------------------------------------------------------
// mish(x) -> bf16 elementwise (conditioning vector c)
// ---------------------------------------------------------------------------
__global__ void mish_bf16_k(const float* __restrict__ in, bf16* __restrict__ out, int n) {
    int i = blockIdx.x * blockDim.x + threadIdx.x;
    if (i < n) out[i] = (bf16)mish_f(in[i]);
}

// ---------------------------------------------------------------------------
// LayerNorm row (768) with DiT modulation: out = ln(x)*(1+scale)+shift (bf16)
// ---------------------------------------------------------------------------
__global__ __launch_bounds__(256)
void ln_mod_k(const float* __restrict__ x, const float* __restrict__ mod,
              int shift_off, int scale_off, bf16* __restrict__ out) {
    __shared__ float red[16];
    const int row = blockIdx.x;       // b*64+s
    const int b   = row >> 6;
    const int tid = threadIdx.x;
    const float* xr = x + (size_t)row * 768;

    float v0 = xr[tid], v1 = xr[tid + 256], v2 = xr[tid + 512];
    float s  = v0 + v1 + v2;
    float s2 = v0 * v0 + v1 * v1 + v2 * v2;
#pragma unroll
    for (int off = 16; off > 0; off >>= 1) {
        s  += __shfl_xor(s,  off, 32);
        s2 += __shfl_xor(s2, off, 32);
    }
    const int wid = tid >> 5, lane = tid & 31;
    if (lane == 0) { red[wid] = s; red[8 + wid] = s2; }
    __syncthreads();
    if (tid == 0) {
        float ts = 0.f, ts2 = 0.f;
        for (int i = 0; i < 8; ++i) { ts += red[i]; ts2 += red[8 + i]; }
        float mean = ts * (1.0f / 768.0f);
        float var  = ts2 * (1.0f / 768.0f) - mean * mean;
        red[0] = mean;
        red[1] = rsqrtf(var + 1e-5f);
    }
    __syncthreads();
    const float mean = red[0], rstd = red[1];
    const float* mb = mod + (size_t)b * 4608;
    bf16* orow = out + (size_t)row * 768;
#pragma unroll
    for (int i = 0; i < 3; ++i) {
        int   c = tid + (i << 8);
        float v = (i == 0 ? v0 : (i == 1 ? v1 : v2));
        float y = (v - mean) * rstd * (1.0f + mb[scale_off + c]) + mb[shift_off + c];
        orow[c] = (bf16)y;
    }
}

// ---------------------------------------------------------------------------
// Double-buffered bf16 WMMA GEMM: D[M][N] = A[M][K] @ BT[N][K]^T + bias.
// BM=BN=128, BK=32; 256 threads = 8 waves (4 M x 2 N), wave tile 32x64.
// Tile staging: TDM > async global->LDS > manual b128 copies.
// ---------------------------------------------------------------------------
enum { EPI_F32 = 0, EPI_BF16 = 1, EPI_GATE = 2, EPI_MISH = 3 };

template <int EPI>
__global__ __launch_bounds__(256)
void gemm_bf16_k(const bf16* __restrict__ A,     // [M][K]
                 const bf16* __restrict__ BT,    // [N][K]
                 const float* __restrict__ bias, // [N]
                 float* __restrict__ outF,
                 bf16*  __restrict__ outH,
                 const float* __restrict__ resid,
                 const float* __restrict__ modbuf,
                 int gate_off, int N, int K) {
    __shared__ alignas(16) bf16 As[2][128][32];
    __shared__ alignas(16) bf16 Bs[2][128][32];

    const int tid  = threadIdx.x;
    const int lane = tid & 31;
    const int wid  = tid >> 5;
    const int wm   = wid & 3;
    const int wn   = wid >> 2;
    const int kh   = lane >> 4;
    const int nl   = lane & 15;

    const int mBlk = blockIdx.y * 128;
    const int nBlk = blockIdx.x * 128;

    auto stage = [&](int buf, int k0) {
#if defined(USE_TDM)
        if (wid == 0) {
            tdm_load_tile_bf16(lds_off_of(&As[buf][0][0]),
                               &A[(size_t)mBlk * K + k0], 32, 128, (unsigned long long)K);
            tdm_load_tile_bf16(lds_off_of(&Bs[buf][0][0]),
                               &BT[(size_t)nBlk * K + k0], 32, 128, (unsigned long long)K);
        }
#elif defined(USE_ASYNC)
#pragma unroll
        for (int i = 0; i < 2; ++i) {
            int idx = tid + (i << 8);
            int row = idx >> 2, seg = (idx & 3) << 3;
            async_b128(&A [(size_t)(mBlk + row) * K + k0 + seg], &As[buf][row][seg]);
            async_b128(&BT[(size_t)(nBlk + row) * K + k0 + seg], &Bs[buf][row][seg]);
        }
#else
#pragma unroll
        for (int i = 0; i < 2; ++i) {
            int idx = tid + (i << 8);
            int row = idx >> 2, seg = (idx & 3) << 3;
            *(U4*)&As[buf][row][seg] = *(const U4*)&A [(size_t)(mBlk + row) * K + k0 + seg];
            *(U4*)&Bs[buf][row][seg] = *(const U4*)&BT[(size_t)(nBlk + row) * K + k0 + seg];
        }
#endif
    };
    auto stage_wait = [&]() {
#if defined(USE_TDM)
        if (wid == 0) tdm_wait0();
#elif defined(USE_ASYNC)
        async_wait0();
#endif
    };

    v8f acc[2][4];
#pragma unroll
    for (int i = 0; i < 2; ++i)
#pragma unroll
        for (int j = 0; j < 4; ++j) acc[i][j] = zero_v8f();

    const int stages = K >> 5;
    stage(0, 0);
    stage_wait();
    __syncthreads();

    for (int kt = 0; kt < stages; ++kt) {
        const int cur = kt & 1;
        if (kt + 1 < stages) stage(cur ^ 1, (kt + 1) << 5);   // prefetch next tile

        FragB16 af[2];
#pragma unroll
        for (int i = 0; i < 2; ++i) {
            int mrow = (wm << 5) + (i << 4) + nl;
            af[i].q[0] = *(const U4*)&As[cur][mrow][(kh << 3)];
            af[i].q[1] = *(const U4*)&As[cur][mrow][16 + (kh << 3)];
        }
        FragB16 bfg[4];
#pragma unroll
        for (int j = 0; j < 4; ++j) {
            int nrow = (wn << 6) + (j << 4) + nl;
            bfg[j].q[0] = *(const U4*)&Bs[cur][nrow][(kh << 3)];
            bfg[j].q[1] = *(const U4*)&Bs[cur][nrow][16 + (kh << 3)];
        }
#pragma unroll
        for (int i = 0; i < 2; ++i)
#pragma unroll
            for (int j = 0; j < 4; ++j)
                acc[i][j] = __builtin_amdgcn_wmma_f32_16x16x32_bf16(
                    false, af[i].v, false, bfg[j].v, (short)0, acc[i][j], false, false);

        if (kt + 1 < stages) {
            stage_wait();
            __syncthreads();
        }
    }

    // epilogue: C layout row = r + 8*kh, col = nl
#pragma unroll
    for (int i = 0; i < 2; ++i) {
#pragma unroll
        for (int j = 0; j < 4; ++j) {
            int   col = nBlk + (wn << 6) + (j << 4) + nl;
            float bv  = bias[col];
#pragma unroll
            for (int r = 0; r < 8; ++r) {
                int    row = mBlk + (wm << 5) + (i << 4) + (kh << 3) + r;
                float  v   = acc[i][j][r] + bv;
                size_t o   = (size_t)row * N + col;
                if      (EPI == EPI_F32)  outF[o] = v;
                else if (EPI == EPI_BF16) outH[o] = (bf16)v;
                else if (EPI == EPI_MISH) outH[o] = (bf16)mish_f(v);
                else {
                    float g = modbuf[(size_t)(row >> 6) * 4608 + gate_off + col];
                    outF[o] = resid[o] + g * v;
                }
            }
        }
    }
}

// ---------------------------------------------------------------------------
// Attention per (b,h): S=64, head_dim=64. 128 threads = 4 waves.
// q/k staged via TDM/async; v transposed manually; WMMA for both GEMMs.
// ---------------------------------------------------------------------------
__global__ __launch_bounds__(128)
void attn_k(const bf16* __restrict__ qkv, bf16* __restrict__ outb) {
    __shared__ alignas(16) bf16 qs[64][64];
    __shared__ alignas(16) bf16 ks[64][64];
    __shared__ alignas(16) bf16 vt[64][64];
    __shared__ alignas(16) bf16 ps[64][64];

    const int b    = blockIdx.x / 12;
    const int h    = blockIdx.x % 12;
    const int tid  = threadIdx.x;
    const int lane = tid & 31;
    const int w    = tid >> 5;
    const int kh   = lane >> 4, nl = lane & 15;

    const size_t rowBase = (size_t)b * 64;
    const int qc = h * 64, kc = 768 + h * 64, vc = 1536 + h * 64;

#if defined(USE_TDM)
    if (w == 0) {
        tdm_load_tile_bf16(lds_off_of(&qs[0][0]), &qkv[rowBase * 2304 + qc], 64, 64, 2304ull);
        tdm_load_tile_bf16(lds_off_of(&ks[0][0]), &qkv[rowBase * 2304 + kc], 64, 64, 2304ull);
    }
#elif defined(USE_ASYNC)
    for (int i = 0; i < 4; ++i) {
        int idx = tid + i * 128;
        int row = idx >> 3, seg = (idx & 7) << 3;
        const size_t g = (rowBase + row) * 2304;
        async_b128(&qkv[g + qc + seg], &qs[row][seg]);
        async_b128(&qkv[g + kc + seg], &ks[row][seg]);
    }
#else
    for (int i = 0; i < 4; ++i) {
        int idx = tid + i * 128;
        int row = idx >> 3, seg = (idx & 7) << 3;
        const size_t g = (rowBase + row) * 2304;
        *(U4*)&qs[row][seg] = *(const U4*)&qkv[g + qc + seg];
        *(U4*)&ks[row][seg] = *(const U4*)&qkv[g + kc + seg];
    }
#endif
    // v transposed through VGPRs (TDM cannot transpose)
    for (int i = 0; i < 4; ++i) {
        int idx = tid + i * 128;
        int row = idx >> 3, seg = (idx & 7) << 3;
        U4 vv = *(const U4*)&qkv[(rowBase + row) * 2304 + vc + seg];
        const bf16* ve = (const bf16*)&vv;
#pragma unroll
        for (int e = 0; e < 8; ++e) vt[seg + e][row] = ve[e];
    }
#if defined(USE_TDM)
    if (w == 0) tdm_wait0();
#elif defined(USE_ASYNC)
    async_wait0();
#endif
    __syncthreads();

    const int mb = w << 4;

    // scores strip (16x64) via WMMA
    v8f sc[4];
#pragma unroll
    for (int j = 0; j < 4; ++j) sc[j] = zero_v8f();
#pragma unroll
    for (int k0 = 0; k0 < 64; k0 += 32) {
        FragB16 a;
        a.q[0] = *(const U4*)&qs[mb + nl][k0 + (kh << 3)];
        a.q[1] = *(const U4*)&qs[mb + nl][k0 + 16 + (kh << 3)];
#pragma unroll
        for (int j = 0; j < 4; ++j) {
            FragB16 bb;
            bb.q[0] = *(const U4*)&ks[(j << 4) + nl][k0 + (kh << 3)];
            bb.q[1] = *(const U4*)&ks[(j << 4) + nl][k0 + 16 + (kh << 3)];
            sc[j] = __builtin_amdgcn_wmma_f32_16x16x32_bf16(
                false, a.v, false, bb.v, (short)0, sc[j], false, false);
        }
    }

    // softmax: row = r + 8*kh spans 16 lanes x 4 accums
    const float scale = 0.125f;
#pragma unroll
    for (int r = 0; r < 8; ++r) {
        float mx = -3.0e38f;
#pragma unroll
        for (int j = 0; j < 4; ++j) { sc[j][r] *= scale; mx = fmaxf(mx, sc[j][r]); }
#pragma unroll
        for (int off = 1; off < 16; off <<= 1) mx = fmaxf(mx, __shfl_xor(mx, off, 32));
        float sum = 0.f;
#pragma unroll
        for (int j = 0; j < 4; ++j) { float e = expf(sc[j][r] - mx); sc[j][r] = e; sum += e; }
#pragma unroll
        for (int off = 1; off < 16; off <<= 1) sum += __shfl_xor(sum, off, 32);
        float inv = 1.0f / sum;
#pragma unroll
        for (int j = 0; j < 4; ++j)
            ps[mb + (kh << 3) + r][(j << 4) + nl] = (bf16)(sc[j][r] * inv);
    }

    // out = P @ V
#pragma unroll
    for (int j = 0; j < 4; ++j) sc[j] = zero_v8f();
#pragma unroll
    for (int t0 = 0; t0 < 64; t0 += 32) {
        FragB16 a;
        a.q[0] = *(const U4*)&ps[mb + nl][t0 + (kh << 3)];
        a.q[1] = *(const U4*)&ps[mb + nl][t0 + 16 + (kh << 3)];
#pragma unroll
        for (int j = 0; j < 4; ++j) {
            FragB16 bb;
            bb.q[0] = *(const U4*)&vt[(j << 4) + nl][t0 + (kh << 3)];
            bb.q[1] = *(const U4*)&vt[(j << 4) + nl][t0 + 16 + (kh << 3)];
            sc[j] = __builtin_amdgcn_wmma_f32_16x16x32_bf16(
                false, a.v, false, bb.v, (short)0, sc[j], false, false);
        }
    }
#pragma unroll
    for (int j = 0; j < 4; ++j) {
        int col = h * 64 + (j << 4) + nl;
#pragma unroll
        for (int r = 0; r < 8; ++r) {
            int s = mb + (kh << 3) + r;
            outb[(rowBase + s) * 768 + col] = (bf16)sc[j][r];
        }
    }
}

// ---------------------------------------------------------------------------
// Host-side orchestration
// ---------------------------------------------------------------------------
extern "C" void kernel_launch(void* const* d_in, const int* in_sizes, int n_in,
                              void* d_out, int out_size, void* d_ws, size_t ws_size,
                              hipStream_t stream) {
    const float* x     = (const float*)d_in[0];
    const float* c     = (const float*)d_in[1];
    const float* W_mod = (const float*)d_in[2];
    const float* b_mod = (const float*)d_in[3];
    const float* W_qkv = (const float*)d_in[4];
    const float* b_qkv = (const float*)d_in[5];
    const float* W_out = (const float*)d_in[6];
    const float* b_out = (const float*)d_in[7];
    const float* W_f1  = (const float*)d_in[8];
    const float* b_f1  = (const float*)d_in[9];
    const float* W_f2  = (const float*)d_in[10];
    const float* b_f2  = (const float*)d_in[11];
    float* out = (float*)d_out;

    const int D = 768, HID = 3072, Bc = 1024, M = 65536;

    char* ws = (char*)d_ws;
    size_t off = 0;
    auto alloc = [&](size_t bytes) -> void* {
        void* p = ws + off;
        off = (off + bytes + 255) & ~(size_t)255;
        return p;
    };
    bf16*  cmish  = (bf16*)alloc((size_t)Bc * D * 2);
    bf16*  WmodT  = (bf16*)alloc((size_t)6 * D * D * 2);
    bf16*  WqkvT  = (bf16*)alloc((size_t)3 * D * D * 2);
    bf16*  WoutT  = (bf16*)alloc((size_t)D * D * 2);
    bf16*  Wf1T   = (bf16*)alloc((size_t)HID * D * 2);
    bf16*  Wf2T   = (bf16*)alloc((size_t)D * HID * 2);
    float* modbuf = (float*)alloc((size_t)Bc * 6 * D * 4);
    bf16*  h1     = (bf16*)alloc((size_t)M * D * 2);
    bf16*  qkvbuf = (bf16*)alloc((size_t)M * 3 * D * 2);
    bf16*  attnb  = (bf16*)alloc((size_t)M * D * 2);
    float* x1     = (float*)alloc((size_t)M * D * 4);
    bf16*  h2     = (bf16*)alloc((size_t)M * D * 2);
    bf16*  f1buf  = (bf16*)alloc((size_t)M * HID * 2);

    auto tgrid = [](size_t total) { return dim3((unsigned)((total + 255) / 256)); };

    transpose_cast_k<<<tgrid((size_t)D * 6 * D), 256, 0, stream>>>(W_mod, WmodT, D, 6 * D);
    transpose_cast_k<<<tgrid((size_t)D * 3 * D), 256, 0, stream>>>(W_qkv, WqkvT, D, 3 * D);
    transpose_cast_k<<<tgrid((size_t)D * D),     256, 0, stream>>>(W_out, WoutT, D, D);
    transpose_cast_k<<<tgrid((size_t)D * HID),   256, 0, stream>>>(W_f1,  Wf1T,  D, HID);
    transpose_cast_k<<<tgrid((size_t)HID * D),   256, 0, stream>>>(W_f2,  Wf2T,  HID, D);

    mish_bf16_k<<<tgrid((size_t)Bc * D), 256, 0, stream>>>(c, cmish, Bc * D);

    gemm_bf16_k<EPI_F32><<<dim3(6 * D / 128, Bc / 128), 256, 0, stream>>>(
        cmish, WmodT, b_mod, modbuf, nullptr, nullptr, nullptr, 0, 6 * D, D);

    ln_mod_k<<<dim3(M), 256, 0, stream>>>(x, modbuf, 0, D, h1);

    gemm_bf16_k<EPI_BF16><<<dim3(3 * D / 128, M / 128), 256, 0, stream>>>(
        h1, WqkvT, b_qkv, nullptr, qkvbuf, nullptr, nullptr, 0, 3 * D, D);

    attn_k<<<dim3(Bc * 12), 128, 0, stream>>>(qkvbuf, attnb);

    gemm_bf16_k<EPI_GATE><<<dim3(D / 128, M / 128), 256, 0, stream>>>(
        attnb, WoutT, b_out, x1, nullptr, x, modbuf, 2 * D, D, D);

    ln_mod_k<<<dim3(M), 256, 0, stream>>>(x1, modbuf, 3 * D, 4 * D, h2);

    gemm_bf16_k<EPI_MISH><<<dim3(HID / 128, M / 128), 256, 0, stream>>>(
        h2, Wf1T, b_f1, nullptr, f1buf, nullptr, nullptr, 0, HID, D);

    gemm_bf16_k<EPI_GATE><<<dim3(D / 128, M / 128), 256, 0, stream>>>(
        f1buf, Wf2T, b_f2, out, nullptr, x1, modbuf, 5 * D, D, HID);

    (void)in_sizes; (void)n_in; (void)out_size; (void)ws_size;
}
